// MemCell_824633721125
// MI455X (gfx1250) — matrix operand order repaired and checked by reading, not compiled, not run on hardware
//
#include <hip/hip_runtime.h>
#include <hip/hip_bf16.h>
#include <math.h>

typedef __attribute__((ext_vector_type(16))) __bf16 v16bf;
typedef __attribute__((ext_vector_type(8)))  float  v8f;

#define B_N   4096
#define J_N   32
#define E_N   256
#define ROWS  (B_N * J_N)   // 131072 (b,j) rows of length E

// ---------------------------------------------------------------------------
// Pre-kernel: repack U (f32 [E,E], row f, col e) into bf16 WMMA B-fragments.
// B-matrix (32x16 bf16) layout: lane l holds N = l&15, element i holds
// K = 8*(l>>4) + (i&7) + 16*(i>>3).  B(k,n) = U[n_glob, k_glob]  (since cand = s @ U^T).
// Linear layout: ub[ ((kt*16+nt)*32 + lane)*16 + i ]
// ---------------------------------------------------------------------------
__global__ void pack_u_kernel(const float* __restrict__ U, __bf16* __restrict__ ub) {
    int idx  = blockIdx.x * 256 + threadIdx.x;   // 65536 elements
    int i    = idx & 15;
    int lane = (idx >> 4) & 31;
    int nt   = (idx >> 9) & 15;
    int kt   = idx >> 13;
    int kg   = kt * 32 + 8 * (lane >> 4) + (i & 7) + 16 * (i >> 3);
    int ng   = nt * 16 + (lane & 15);
    ub[idx] = (__bf16)U[ng * E_N + kg];
}

// vk[j,f] = sum_e V[f,e] * keys[j,e]   (tiny: 32x256 output, one block)
__global__ void vk_kernel(const float* __restrict__ V, const float* __restrict__ keys,
                          float* __restrict__ vk) {
    __shared__ float ks[J_N * E_N];
    int f = threadIdx.x;
    for (int t = f; t < J_N * E_N; t += 256) ks[t] = keys[t];
    __syncthreads();
    float acc[J_N];
    #pragma unroll
    for (int j = 0; j < J_N; j++) acc[j] = 0.f;
    const float* vr = V + f * E_N;
    for (int e = 0; e < E_N; e++) {
        float v = vr[e];
        #pragma unroll
        for (int j = 0; j < J_N; j++) acc[j] = fmaf(v, ks[j * E_N + e], acc[j]);
    }
    for (int j = 0; j < J_N; j++) vk[j * E_N + f] = acc[j];
}

// xk[b,j] = x[b,:] . keys[j,:]
__global__ void xk_kernel(const float* __restrict__ x, const float* __restrict__ keys,
                          float* __restrict__ xk) {
    __shared__ float ks[J_N * E_N];
    int tid = threadIdx.x;
    for (int t = tid; t < J_N * E_N; t += 256) ks[t] = keys[t];
    __syncthreads();
    int r = blockIdx.x * 256 + tid;
    const float* xr = x + (size_t)(r >> 5) * E_N;
    const float* kr = ks + (r & 31) * E_N;
    float a = 0.f;
    for (int e = 0; e < E_N; e++) a = fmaf(xr[e], kr[e], a);
    xk[r] = a;
}

// wxb[b,f] = sum_e W[f,e]*x[b,e] + bias[f]   (16 batch rows per block to reuse W)
__global__ void wxb_kernel(const float* __restrict__ x, const float* __restrict__ W,
                           const float* __restrict__ bias, float* __restrict__ wxb) {
    __shared__ float xs[16][E_N];
    int f  = threadIdx.x;
    int b0 = blockIdx.x * 16;
    for (int i = 0; i < 16; i++) xs[i][f] = x[(size_t)(b0 + i) * E_N + f];
    __syncthreads();
    float acc[16];
    float bv = bias[f];
    #pragma unroll
    for (int i = 0; i < 16; i++) acc[i] = bv;
    const float* wr = W + f * E_N;
    for (int e = 0; e < E_N; e++) {
        float w = wr[e];
        #pragma unroll
        for (int i = 0; i < 16; i++) acc[i] = fmaf(w, xs[i][e], acc[i]);
    }
    for (int i = 0; i < 16; i++) wxb[(size_t)(b0 + i) * E_N + f] = acc[i];
}

// ---------------------------------------------------------------------------
// Main kernel: 8 wave32s per block = 4 row-groups of 16 rows, with the N=256
// output split across a PAIR of waves (128 cols each) so acc = 8 x v8f = 64
// VGPRs per lane.  K-loop of 8 x v_wmma_f32_16x16x32_bf16 per wave.  The
// L2-norm crosses the wave pair via a 512B LDS buffer + one barrier.
// ---------------------------------------------------------------------------
__global__ __launch_bounds__(256, 2)
void memcell_main(const float* __restrict__ x, const float* __restrict__ state,
                  const __bf16* __restrict__ ub, const float* __restrict__ wxb,
                  const float* __restrict__ vk, const float* __restrict__ xk,
                  float* __restrict__ out) {
    const int lane = threadIdx.x & 31;
    const int wave = threadIdx.x >> 5;      // 0..7
    const int grp  = wave >> 1;             // row group 0..3
    const int nh   = wave & 1;              // N half: columns [nh*128, nh*128+128)
    const int h    = lane >> 4;             // half of the wave
    const int ln   = lane & 15;
    const int rowBase = blockIdx.x * 64 + grp * 16;
    const int myRow   = rowBase + ln;       // A-fragment row owned by this lane
    const float* sRow = state + (size_t)myRow * E_N;
    const float* xRow = x + (size_t)(myRow >> 5) * E_N;

    v8f acc[8];
    const v8f zero8 = {0.f, 0.f, 0.f, 0.f, 0.f, 0.f, 0.f, 0.f};
    #pragma unroll
    for (int ntl = 0; ntl < 8; ntl++) acc[ntl] = zero8;

    float dot = 0.f;   // partial of x . s over this lane's K slices

    for (int kt = 0; kt < 8; kt++) {
        // A-matrix 16x32 bf16 layout: lane half h, elem i -> K = kt*32 + i + 8h + 8*(i>=8)
        const int c0 = kt * 32 + 8 * h;
        const int c1 = c0 + 16;
        float4 fv[4], xv[4];
        fv[0] = *(const float4*)(sRow + c0);
        fv[1] = *(const float4*)(sRow + c0 + 4);
        fv[2] = *(const float4*)(sRow + c1);
        fv[3] = *(const float4*)(sRow + c1 + 4);
        xv[0] = *(const float4*)(xRow + c0);
        xv[1] = *(const float4*)(xRow + c0 + 4);
        xv[2] = *(const float4*)(xRow + c1);
        xv[3] = *(const float4*)(xRow + c1 + 4);
        __builtin_prefetch(sRow + c0 + 32, 0, 1);   // global_prefetch_b8 (next K-chunk)

        const float* fp = (const float*)fv;
        const float* xp = (const float*)xv;
        v16bf a;
        #pragma unroll
        for (int i = 0; i < 16; i++) {
            dot = fmaf(fp[i], xp[i], dot);          // gate dot stays full f32
            a[i] = (__bf16)fp[i];
        }

        const v16bf* bp = (const v16bf*)(ub + (size_t)kt * 16 * 32 * 16);
        #pragma unroll
        for (int ntl = 0; ntl < 8; ntl++) {
            v16bf bfrag = bp[(nh * 8 + ntl) * 32 + lane];
            acc[ntl] = __builtin_amdgcn_wmma_f32_16x16x32_bf16(
                false, a, false, bfrag, (short)0, acc[ntl], false, false);
        }
    }

    // gate per row: full dot = this half + other half, then sigmoid
    // (recomputed identically in both waves of the pair -> no sync needed)
    dot += __shfl_xor(dot, 16, 32);
    float g = 1.f / (1.f + __expf(-(dot + xk[myRow])));

    // Epilogue. C/D layout: lane holds (m = v + 8h, n = (nh*8+ntl)*16 + ln).
    __shared__ float nsqbuf[8][2][8];   // [wave][half][v]
    float nsqp[8];
    #pragma unroll
    for (int v = 0; v < 8; v++) {
        const int m = v + 8 * h;
        const int r = rowBase + m;
        const float* wr = wxb + (size_t)(r >> 5) * E_N;
        const float* vr = vk + (size_t)(r & 31) * E_N;
        const float* sr = state + (size_t)r * E_N;
        const float gm = __shfl(g, m, 32);   // gate of row m lives at lane m

        float nsq = 0.f;
        #pragma unroll
        for (int ntl = 0; ntl < 8; ntl++) {
            const int n = (nh * 8 + ntl) * 16 + ln;
            float pre = acc[ntl][v] + wr[n] + vr[n];
            float c   = fmaxf(pre, 0.f);
            float t   = fmaf(gm, c, sr[n]);   // s + gate*cand, s re-read in f32
            acc[ntl][v] = t;                  // keep s_next in place of acc
            nsq = fmaf(t, t, nsq);
        }
        // reduce over the 16 lanes sharing row m (stays within the half)
        nsq += __shfl_xor(nsq, 1, 32);
        nsq += __shfl_xor(nsq, 2, 32);
        nsq += __shfl_xor(nsq, 4, 32);
        nsq += __shfl_xor(nsq, 8, 32);
        nsqp[v] = nsq;                        // this wave's half-row sum of squares
    }

    if (ln == 0) {
        #pragma unroll
        for (int v = 0; v < 8; v++) nsqbuf[wave][h][v] = nsqp[v];
    }
    __syncthreads();

    #pragma unroll
    for (int v = 0; v < 8; v++) {
        const int r = rowBase + v + 8 * h;
        const float total = nsqp[v] + nsqbuf[wave ^ 1][h][v];
        const float scale = 1.f / (sqrtf(total) + 1e-8f);
        float* orow = out + (size_t)r * E_N;
        #pragma unroll
        for (int ntl = 0; ntl < 8; ntl++)
            orow[(nh * 8 + ntl) * 16 + ln] = acc[ntl][v] * scale;
    }
}

// ---------------------------------------------------------------------------
extern "C" void kernel_launch(void* const* d_in, const int* in_sizes, int n_in,
                              void* d_out, int out_size, void* d_ws, size_t ws_size,
                              hipStream_t stream) {
    const float* x     = (const float*)d_in[0];
    const float* state = (const float*)d_in[1];
    const float* keys  = (const float*)d_in[2];
    const float* U     = (const float*)d_in[3];
    const float* V     = (const float*)d_in[4];
    const float* W     = (const float*)d_in[5];
    const float* bias  = (const float*)d_in[6];
    float* out = (float*)d_out;

    char* ws = (char*)d_ws;
    float*  wxb = (float*)(ws);                                   // 4 MB
    float*  xk  = (float*)(ws + 4u * 1024 * 1024);                // 512 KB
    float*  vkp = (float*)(ws + 4u * 1024 * 1024 + 512u * 1024);  // 32 KB
    __bf16* ub  = (__bf16*)(ws + 4u * 1024 * 1024 + 512u * 1024 + 32u * 1024); // 128 KB

    pack_u_kernel<<<256, 256, 0, stream>>>(U, ub);
    vk_kernel<<<1, 256, 0, stream>>>(V, keys, vkp);
    xk_kernel<<<ROWS / 256, 256, 0, stream>>>(x, keys, xk);
    wxb_kernel<<<B_N / 16, 256, 0, stream>>>(x, W, bias, wxb);
    memcell_main<<<ROWS / 64, 256, 0, stream>>>(x, state, ub, wxb, vkp, xk, out);
}
